// MultiHeadAttention_78597901517366
// MI455X (gfx1250) — matrix operand order repaired and checked by reading, not compiled, not run on hardware
//
#include <hip/hip_runtime.h>

typedef __attribute__((ext_vector_type(16))) _Float16 v16h;
typedef __attribute__((ext_vector_type(8)))  _Float16 v8h;
typedef __attribute__((ext_vector_type(8)))  float    v8f;

static __device__ __forceinline__ v16h cat16(v8h lo, v8h hi) {
    return __builtin_shufflevector(lo, hi, 0,1,2,3,4,5,6,7,8,9,10,11,12,13,14,15);
}
static __device__ __forceinline__ v8f wmma_f16(v16h a, v16h b, v8f c) {
    return __builtin_amdgcn_wmma_f32_16x16x32_f16(false, a, false, b, (short)0, c, false, false);
}

#define LDA_STRIDE 40   // 128 rows x 32 K halves, padded
#define LDB_STRIDE 40   // 128 cols x 32 K halves, padded (B stored transposed)

// ---------------------------------------------------------------------------
// Kernel 1: QKV projection. Y = x @ W  (M=32768, N=384, K=384), f32 in,
// f16 out. z=0 -> Q (flat), z=1 -> K (flat), z=2 -> V stored transposed
// per-head as [64, 256]. Workgroup tile 128x128, wave tile 32x64.
// ---------------------------------------------------------------------------
__global__ __launch_bounds__(256) void mha_qkv_gemm(
    const float* __restrict__ x, const float* __restrict__ Wq,
    const float* __restrict__ Wk, const float* __restrict__ Wv,
    _Float16* __restrict__ qb, _Float16* __restrict__ kb,
    _Float16* __restrict__ vtb)
{
    __shared__ __align__(16) _Float16 ldsA[128 * LDA_STRIDE];
    __shared__ __align__(16) _Float16 ldsB[128 * LDB_STRIDE];

    const int z    = blockIdx.z;
    const float* W = (z == 0) ? Wq : (z == 1) ? Wk : Wv;
    const int m0   = blockIdx.x * 128;
    const int n0   = blockIdx.y * 128;
    const int tid  = threadIdx.x;
    const int lane = tid & 31;
    const int wid  = tid >> 5;
    const int wm   = (wid >> 1) * 32;   // 0,32,64,96
    const int wn   = (wid & 1) * 64;    // 0,64
    const int l16  = lane & 15;
    const int g    = lane >> 4;

    // staging source pointers (advance by 32 floats per K step)
    const int arow = tid >> 1;
    const int acc0 = (tid & 1) * 16;
    const float* asrc = x + (size_t)(m0 + arow) * 384 + acc0;
    const int bkk  = tid & 31;
    const int bnn0 = (tid >> 5) * 16;
    const float* bsrc = W + (size_t)bkk * 384 + n0 + bnn0;

    v8f acc[2][4] = {};

    for (int k0 = 0; k0 < 384; k0 += 32) {
        if (k0 + 32 < 384) {   // prefetch next K slice into WGP-level caches
            __builtin_prefetch(asrc + 32, 0, 3);
            __builtin_prefetch(bsrc + (size_t)32 * 384, 0, 3);
        }
        // stage A: 128x32 tile, f32 -> f16 (2 threads per row, 16 elems each)
        {
            _Float16* dst = &ldsA[arow * LDA_STRIDE + acc0];
            #pragma unroll
            for (int j = 0; j < 16; ++j) dst[j] = (_Float16)asrc[j];
        }
        // stage B transposed: ldsB[n][kk] = W[k0+kk][n0+n]
        {
            #pragma unroll
            for (int j = 0; j < 16; ++j)
                ldsB[(bnn0 + j) * LDB_STRIDE + bkk] = (_Float16)bsrc[j];
        }
        __syncthreads();

        v16h a[2], b[4];
        #pragma unroll
        for (int mi = 0; mi < 2; ++mi) {
            const int row = wm + mi * 16 + l16;
            v8h lo = *(const v8h*)&ldsA[row * LDA_STRIDE + g * 16];
            v8h hi = *(const v8h*)&ldsA[row * LDA_STRIDE + g * 16 + 8];
            a[mi] = cat16(lo, hi);
        }
        #pragma unroll
        for (int ni = 0; ni < 4; ++ni) {
            const int col = wn + ni * 16 + l16;
            v8h lo = *(const v8h*)&ldsB[col * LDB_STRIDE + g * 16];
            v8h hi = *(const v8h*)&ldsB[col * LDB_STRIDE + g * 16 + 8];
            b[ni] = cat16(lo, hi);
        }
        #pragma unroll
        for (int mi = 0; mi < 2; ++mi)
            #pragma unroll
            for (int ni = 0; ni < 4; ++ni)
                acc[mi][ni] = wmma_f16(a[mi], b[ni], acc[mi][ni]);
        __syncthreads();

        asrc += 32;
        bsrc += (size_t)32 * 384;
    }

    // epilogue: z-branch hoisted out of the store loops
    if (z != 2) {
        _Float16* dst = (z == 0) ? qb : kb;
        #pragma unroll
        for (int mi = 0; mi < 2; ++mi)
        #pragma unroll
        for (int ni = 0; ni < 4; ++ni)
        #pragma unroll
        for (int r = 0; r < 8; ++r) {
            const int m = m0 + wm + mi * 16 + g * 8 + r;
            const int n = n0 + wn + ni * 16 + l16;
            dst[(size_t)m * 384 + n] = (_Float16)acc[mi][ni][r];
        }
    } else {
        #pragma unroll
        for (int mi = 0; mi < 2; ++mi)
        #pragma unroll
        for (int ni = 0; ni < 4; ++ni)
        #pragma unroll
        for (int r = 0; r < 8; ++r) {
            const int m = m0 + wm + mi * 16 + g * 8 + r;
            const int n = n0 + wn + ni * 16 + l16;
            // V^T per head: [64 dims][256 tokens]
            const int bidx = m >> 8, t = m & 255;
            const int flat = t * 384 + n;
            const int h = flat >> 14, o = flat & 16383;
            vtb[(size_t)(bidx * 6 + h) * 16384 + (size_t)(o & 63) * 256 + (o >> 6)]
                = (_Float16)acc[mi][ni][r];
        }
    }
}

// ---------------------------------------------------------------------------
// Kernel 2: causal attention per (b,h). Q,K row-major [256,64] f16,
// Vt [64,256] f16, output O row-major [256,64] f16. Flash-style online
// softmax, 16-row query tiles per wave, 32-key blocks.
// ---------------------------------------------------------------------------
__global__ __launch_bounds__(256) void mha_attn(
    const _Float16* __restrict__ qb, const _Float16* __restrict__ kb,
    const _Float16* __restrict__ vtb, _Float16* __restrict__ ob)
{
    __shared__ __align__(16) _Float16 ldsP[8][16 * 32]; // per-wave P tile

    const int bh = blockIdx.x;  // 0..767
    const _Float16* Q  = qb  + (size_t)bh * 16384;
    const _Float16* K  = kb  + (size_t)bh * 16384;
    const _Float16* Vt = vtb + (size_t)bh * 16384;
    _Float16*       O  = ob  + (size_t)bh * 16384;

    const int lane = threadIdx.x & 31;
    const int wid  = threadIdx.x >> 5;
    const int l16  = lane & 15;
    const int g    = lane >> 4;
    const float scale = 0.05103103630798288f;  // 384^-0.5
    _Float16* pP = &ldsP[wid][0];

    for (int qt = wid; qt < 16; qt += 8) {
        const int qr0 = qt * 16;

        // Q A-fragments for the two 32-wide k-chunks of hs=64
        v16h aQ[2];
        #pragma unroll
        for (int c = 0; c < 2; ++c) {
            const _Float16* qrow = Q + (size_t)(qr0 + l16) * 64 + c * 32;
            v8h lo = *(const v8h*)(qrow + g * 16);
            v8h hi = *(const v8h*)(qrow + g * 16 + 8);
            aQ[c] = cat16(lo, hi);
        }

        v8f o_acc[4] = {};
        float mstat[8], lstat[8];
        #pragma unroll
        for (int r = 0; r < 8; ++r) { mstat[r] = -1e30f; lstat[r] = 0.0f; }

        const int nkb = (qr0 + 15) / 32 + 1;  // causal early exit
        for (int jb = 0; jb < nkb; ++jb) {
            const int kb0 = jb * 32;

            // S = Q @ K^T for 32 keys: two 16x16 tiles
            v8f s[2] = {};
            #pragma unroll
            for (int st = 0; st < 2; ++st) {
                const int key = kb0 + st * 16 + l16;
                #pragma unroll
                for (int c = 0; c < 2; ++c) {
                    const _Float16* krow = K + (size_t)key * 64 + c * 32 + g * 16;
                    v8h lo = *(const v8h*)krow;
                    v8h hi = *(const v8h*)(krow + 8);
                    s[st] = wmma_f16(aQ[c], cat16(lo, hi), s[st]);
                }
            }

            // mask + online softmax update
            float p0[8], p1[8];
            #pragma unroll
            for (int r = 0; r < 8; ++r) {
                const int row = qr0 + g * 8 + r;
                float s0 = s[0][r] * scale;
                float s1 = s[1][r] * scale;
                if (kb0 + l16      > row) s0 = -1e30f;
                if (kb0 + 16 + l16 > row) s1 = -1e30f;
                float mx = fmaxf(s0, s1);
                #pragma unroll
                for (int off = 1; off < 16; off <<= 1)
                    mx = fmaxf(mx, __shfl_xor(mx, off, 16));
                const float mnew = fmaxf(mstat[r], mx);
                const float fac  = __expf(mstat[r] - mnew);
                const float e0   = __expf(s0 - mnew);
                const float e1   = __expf(s1 - mnew);
                float rs = e0 + e1;
                #pragma unroll
                for (int off = 1; off < 16; off <<= 1)
                    rs += __shfl_xor(rs, off, 16);
                lstat[r] = lstat[r] * fac + rs;
                mstat[r] = mnew;
                p0[r] = e0; p1[r] = e1;
                #pragma unroll
                for (int ni = 0; ni < 4; ++ni) o_acc[ni][r] *= fac;
            }

            // P through LDS to re-layout C-frag -> A-frag (wave-local)
            #pragma unroll
            for (int r = 0; r < 8; ++r) {
                pP[(g * 8 + r) * 32 + l16]      = (_Float16)p0[r];
                pP[(g * 8 + r) * 32 + 16 + l16] = (_Float16)p1[r];
            }
            asm volatile("s_wait_dscnt 0" ::: "memory");
            v8h plo = *(const v8h*)&pP[l16 * 32 + g * 16];
            v8h phi = *(const v8h*)&pP[l16 * 32 + g * 16 + 8];
            const v16h aP = cat16(plo, phi);
            asm volatile("" ::: "memory");

            // O += P @ V   (B-frags contiguous thanks to V^T layout)
            #pragma unroll
            for (int ni = 0; ni < 4; ++ni) {
                const _Float16* vrow =
                    Vt + (size_t)(ni * 16 + l16) * 256 + kb0 + g * 16;
                v8h blo = *(const v8h*)vrow;
                v8h bhi = *(const v8h*)(vrow + 8);
                o_acc[ni] = wmma_f16(aP, cat16(blo, bhi), o_acc[ni]);
            }
        }

        // normalize + store f16
        #pragma unroll
        for (int r = 0; r < 8; ++r) {
            const float inv = 1.0f / lstat[r];
            const int row = qr0 + g * 8 + r;
            #pragma unroll
            for (int ni = 0; ni < 4; ++ni)
                O[(size_t)row * 64 + ni * 16 + l16] =
                    (_Float16)(o_acc[ni][r] * inv);
        }
    }
}

// ---------------------------------------------------------------------------
// Kernel 3: output projection. out = O @ Wp + bp (f16 A, f32 W/out).
// Workgroup tile 128x128, wave tile 32x64. A tile staged to LDS with
// GLOBAL_LOAD_ASYNC_TO_LDS_B128 (ASYNCcnt path, no VGPR round-trip).
// ---------------------------------------------------------------------------
__global__ __launch_bounds__(256) void mha_out_proj(
    const _Float16* __restrict__ A, const float* __restrict__ Wp,
    const float* __restrict__ bp, float* __restrict__ out)
{
    __shared__ __align__(16) _Float16 ldsA[128 * LDA_STRIDE];
    __shared__ __align__(16) _Float16 ldsB[128 * LDB_STRIDE];

    const int m0   = blockIdx.x * 128;
    const int n0   = blockIdx.y * 128;
    const int tid  = threadIdx.x;
    const int lane = tid & 31;
    const int wid  = tid >> 5;
    const int wm   = (wid >> 1) * 32;
    const int wn   = (wid & 1) * 64;
    const int l16  = lane & 15;
    const int g    = lane >> 4;

    const int arow = tid >> 1;
    const int acc0 = (tid & 1) * 16;
    const _Float16* asrc = A + (size_t)(m0 + arow) * 384 + acc0;
    const int bkk  = tid & 31;
    const int bnn0 = (tid >> 5) * 16;
    const float* bsrc = Wp + (size_t)bkk * 384 + n0 + bnn0;

    // LDS byte offset of this thread's A-tile slot (generic LDS addresses
    // truncate to the LDS offset in the low 32 bits).
    const unsigned ldsA_off =
        (unsigned)(uintptr_t)&ldsA[arow * LDA_STRIDE + acc0];

    v8f acc[2][4] = {};

    for (int k0 = 0; k0 < 384; k0 += 32) {
        if (k0 + 32 < 384) {
            __builtin_prefetch(asrc + 32, 0, 3);
            __builtin_prefetch(bsrc + (size_t)32 * 384, 0, 3);
        }
        // stage A: async global -> LDS copy, 32 bytes per thread
        {
            const unsigned long long ga = (unsigned long long)(uintptr_t)asrc;
            asm volatile(
                "global_load_async_to_lds_b128 %0, %1, off\n\t"
                "global_load_async_to_lds_b128 %0, %1, off offset:16"
                :: "v"(ldsA_off), "v"(ga) : "memory");
        }
        // stage B transposed with f32 -> f16 conversion
        {
            #pragma unroll
            for (int j = 0; j < 16; ++j)
                ldsB[(bnn0 + j) * LDB_STRIDE + bkk] = (_Float16)bsrc[j];
        }
        asm volatile("s_wait_asynccnt 0" ::: "memory");
        __syncthreads();

        v16h a[2], b[4];
        #pragma unroll
        for (int mi = 0; mi < 2; ++mi) {
            const int row = wm + mi * 16 + l16;
            v8h lo = *(const v8h*)&ldsA[row * LDA_STRIDE + g * 16];
            v8h hi = *(const v8h*)&ldsA[row * LDA_STRIDE + g * 16 + 8];
            a[mi] = cat16(lo, hi);
        }
        #pragma unroll
        for (int ni = 0; ni < 4; ++ni) {
            const int col = wn + ni * 16 + l16;
            v8h lo = *(const v8h*)&ldsB[col * LDB_STRIDE + g * 16];
            v8h hi = *(const v8h*)&ldsB[col * LDB_STRIDE + g * 16 + 8];
            b[ni] = cat16(lo, hi);
        }
        #pragma unroll
        for (int mi = 0; mi < 2; ++mi)
            #pragma unroll
            for (int ni = 0; ni < 4; ++ni)
                acc[mi][ni] = wmma_f16(a[mi], b[ni], acc[mi][ni]);
        __syncthreads();

        asrc += 32;
        bsrc += (size_t)32 * 384;
    }

    #pragma unroll
    for (int mi = 0; mi < 2; ++mi)
    #pragma unroll
    for (int ni = 0; ni < 4; ++ni)
    #pragma unroll
    for (int r = 0; r < 8; ++r) {
        const int m = m0 + wm + mi * 16 + g * 8 + r;
        const int n = n0 + wn + ni * 16 + l16;
        out[(size_t)m * 384 + n] = acc[mi][ni][r] + bp[n];
    }
}

// ---------------------------------------------------------------------------
// Launch: QKV -> attention -> projection, intermediates in d_ws (f16).
// ws layout: q | k | v^T | o, each 32768*384 halves = 25,165,824 B.
// ---------------------------------------------------------------------------
extern "C" void kernel_launch(void* const* d_in, const int* in_sizes, int n_in,
                              void* d_out, int out_size, void* d_ws, size_t ws_size,
                              hipStream_t stream) {
    const float* x  = (const float*)d_in[0];
    const float* Wk = (const float*)d_in[1];
    const float* Wq = (const float*)d_in[2];
    const float* Wv = (const float*)d_in[3];
    const float* Wp = (const float*)d_in[4];
    const float* bp = (const float*)d_in[5];
    float* out = (float*)d_out;

    const size_t SEG = (size_t)32768 * 384 * sizeof(_Float16);  // 25,165,824 B
    char* ws = (char*)d_ws;
    _Float16* qb  = (_Float16*)(ws);
    _Float16* kb  = (_Float16*)(ws + SEG);
    _Float16* vtb = (_Float16*)(ws + 2 * SEG);
    _Float16* ob  = (_Float16*)(ws + 3 * SEG);

    mha_qkv_gemm<<<dim3(256, 3, 3), 256, 0, stream>>>(x, Wq, Wk, Wv, qb, kb, vtb);
    mha_attn<<<dim3(768), 256, 0, stream>>>(qb, kb, vtb, ob);
    mha_out_proj<<<dim3(256, 3), 256, 0, stream>>>(ob, Wp, bp, out);
}
